// MulLeaBlock_15573551415935
// MI455X (gfx1250) — matrix operand order repaired and checked by reading, not compiled, not run on hardware
//
#include <hip/hip_runtime.h>
#include <hip/hip_bf16.h>
#include <stdint.h>

typedef __attribute__((ext_vector_type(16))) _Float16 v16h;
typedef __attribute__((ext_vector_type(8)))  float    v8f;

constexpr int Bn = 4;     // batch
constexpr int C  = 256;   // channels
constexpr int CI = 128;   // c/2
constexpr int NP = 4096;  // h*w

__device__ inline v8f wmma_f16(v16h a, v16h b, v8f c) {
  // (neg_a, A, neg_b, B, c_mod, C, reuse_a, reuse_b)
  return __builtin_amdgcn_wmma_f32_16x16x32_f16(false, a, false, b, (short)0, c,
                                                false, false);
}

// A-fragment (16x32 f16), A[m,k] = p[m*ld + k].  Per ISA table a lane's 16
// halves are two contiguous 8-half runs -> two ds_load_b128.
__device__ inline v16h frag_a_rm(const _Float16* p, int ld) {
  int lane = threadIdx.x & 31;
  int m = lane & 15, hh = lane >> 4;
  union { uint4 q[2]; v16h v; } f;
  f.q[0] = *(const uint4*)(p + m * ld + hh * 8);
  f.q[1] = *(const uint4*)(p + m * ld + 16 + hh * 8);
  return f.v;
}
// B-fragment (32x16 f16) from N-major storage, B[k,n] = p[n*ld + k].
// Lane's 16 halves are two contiguous 8-half runs (k = 8hh.., 16+8hh..) -> two ds_load_b128.
__device__ inline v16h frag_b_nm(const _Float16* p, int ld) {
  int lane = threadIdx.x & 31;
  int n = lane & 15, hh = lane >> 4;
  union { uint4 q[2]; v16h v; } f;
  f.q[0] = *(const uint4*)(p + n * ld + hh * 8);
  f.q[1] = *(const uint4*)(p + n * ld + 16 + hh * 8);
  return f.v;
}

__device__ inline void cvt4(_Float16* d, float4 s) {
  d[0] = (_Float16)s.x; d[1] = (_Float16)s.y;
  d[2] = (_Float16)s.z; d[3] = (_Float16)s.w;
}

// ---------------------------------------------------------------------------
// Kernel 1: 1x1 conv as GEMM.  out[n][o] = sum_c W[o][c] * X[b][c][n], f16.
// TRANSPOSE_OUT=1 stores out as [o][n] (used for G) with packed b128 stores.
// Block: 256 thr (8 waves), tile 128n x 128o, K-loop in steps of 32.
// ---------------------------------------------------------------------------
template <int TRANSPOSE_OUT>
__global__ __launch_bounds__(256)
void proj_kernel(const float* __restrict__ W, const float* __restrict__ X,
                 _Float16* __restrict__ Out, int Odim, int Cdim) {
  __shared__ __attribute__((aligned(16))) _Float16 xs[128 * 32];   // X^T [m][k]
  __shared__ __attribute__((aligned(16))) _Float16 wsh[128 * 32];  // W   [n][k]
  int n0 = blockIdx.x * 128, o0 = blockIdx.y * 128, b = blockIdx.z;
  const float* Xb = X + (size_t)b * Cdim * NP;
  _Float16* Ob = Out + (size_t)b * NP * Odim;
  int tid = threadIdx.x, w = tid >> 5, lane = tid & 31;
  v8f zeroV = {};
  v8f acc[8];
#pragma unroll
  for (int t = 0; t < 8; ++t) acc[t] = zeroV;

  for (int c0 = 0; c0 < Cdim; c0 += 32) {
    // X tile: read float4 along n, store transposed [m][k]
    for (int idx = tid; idx < 32 * 32; idx += 256) {
      int k = idx >> 5, mq = idx & 31;
      float4 v = *(const float4*)(Xb + (size_t)(c0 + k) * NP + n0 + mq * 4);
      _Float16 h[4]; cvt4(h, v);
#pragma unroll
      for (int e = 0; e < 4; ++e) xs[(mq * 4 + e) * 32 + k] = h[e];
    }
    // W tile: read float4 along k, packed uint2 store [n][k]
    for (int idx = tid; idx < 128 * 8; idx += 256) {
      int n = idx >> 3, kq = idx & 7;
      float4 v = *(const float4*)(W + (size_t)(o0 + n) * Cdim + c0 + kq * 4);
      union { _Float16 h[4]; uint2 u; } pk; cvt4(pk.h, v);
      *(uint2*)(wsh + n * 32 + kq * 4) = pk.u;
    }
    __syncthreads();
    v16h a = frag_a_rm(xs + (w * 16) * 32, 32);
#pragma unroll
    for (int t = 0; t < 8; ++t) {
      v16h bf = frag_b_nm(wsh + (t * 16) * 32, 32);
      acc[t] = wmma_f16(a, bf, acc[t]);
    }
    __syncthreads();
  }
  int hh = lane >> 4, nn = lane & 15;
  if (TRANSPOSE_OUT) {
    // lane's 8 rows r map to contiguous n -> one b128 store per tile
#pragma unroll
    for (int t = 0; t < 8; ++t) {
      union { _Float16 h[8]; uint4 u; } pk;
#pragma unroll
      for (int r = 0; r < 8; ++r) pk.h[r] = (_Float16)acc[t][r];
      *(uint4*)(Ob + (size_t)(o0 + t * 16 + nn) * NP + n0 + w * 16 + 8 * hh) = pk.u;
    }
  } else {
#pragma unroll
    for (int t = 0; t < 8; ++t)
#pragma unroll
      for (int r = 0; r < 8; ++r) {
        int m = r + 8 * hh;
        Ob[(size_t)(n0 + w * 16 + m) * Odim + o0 + t * 16 + nn] = (_Float16)acc[t][r];
      }
  }
}

// ---------------------------------------------------------------------------
// Kernel 2: column softmax stats. s[i,j] = Kt[i,:] . Q[j,:]; softmax over i.
// Block owns 128 j columns (wave: 16). Q B-fragments are loop-invariant and
// hoisted; the i-loop only streams 4KB Kt tiles + 4 WMMAs.
// ---------------------------------------------------------------------------
__global__ __launch_bounds__(256)
void stats_kernel(const _Float16* __restrict__ Q, const _Float16* __restrict__ Kt,
                  float* __restrict__ mx, float* __restrict__ sm) {
  __shared__ __attribute__((aligned(16))) _Float16 qs[128 * 128];  // 32 KB
  __shared__ __attribute__((aligned(16))) _Float16 ks[16 * 128];   // 4 KB
  int j0 = blockIdx.x * 128, b = blockIdx.y;
  const _Float16* Qb = Q + (size_t)b * NP * CI + (size_t)j0 * CI;
  const _Float16* Kb = Kt + (size_t)b * NP * CI;
  int tid = threadIdx.x, w = tid >> 5, lane = tid & 31;
  {
    const uint4* src = (const uint4*)Qb;
    uint4* dst = (uint4*)qs;
    for (int i = tid; i < (128 * 128) / 8; i += 256) dst[i] = src[i];
  }
  __syncthreads();
  v16h bq[4];  // loop-invariant B fragments (this wave's 16 columns)
#pragma unroll
  for (int kk = 0; kk < 4; ++kk)
    bq[kk] = frag_b_nm(qs + (w * 16) * 128 + kk * 32, 128);

  float m_run = -3.0e38f, l_run = 0.0f;
  for (int it = 0; it < NP / 16; ++it) {
    ((uint4*)ks)[tid] = ((const uint4*)(Kb + (size_t)it * 16 * CI))[tid];
    if (it + 1 < NP / 16)  // -> global_prefetch_b8 for next K-tile
      __builtin_prefetch(Kb + (size_t)(it + 1) * 16 * CI + tid * 8, 0, 1);
    __syncthreads();
    v8f s = {};
#pragma unroll
    for (int kk = 0; kk < 4; ++kk) {
      v16h a = frag_a_rm(ks + kk * 32, 128);
      s = wmma_f16(a, bq[kk], s);
    }
    float tmax = s[0];
#pragma unroll
    for (int r = 1; r < 8; ++r) tmax = fmaxf(tmax, s[r]);
    float nm = fmaxf(m_run, tmax);
    float l = l_run * __expf(m_run - nm);
#pragma unroll
    for (int r = 0; r < 8; ++r) l += __expf(s[r] - nm);
    m_run = nm; l_run = l;
    __syncthreads();
  }
  // lanes L and L+16 own the same column (different i rows): merge (wave32)
  float m2 = __shfl_xor(m_run, 16, 32);
  float l2 = __shfl_xor(l_run, 16, 32);
  float nm = fmaxf(m_run, m2);
  float lt = l_run * __expf(m_run - nm) + l2 * __expf(m2 - nm);
  if (lane < 16) {
    mx[(size_t)b * NP + j0 + w * 16 + lane] = nm;
    sm[(size_t)b * NP + j0 + w * 16 + lane] = lt;
  }
}

// ---------------------------------------------------------------------------
// Kernel 3: out[i,:] = sum_j exp(s[i,j]-mx[j])/sm[j] * G[j,:].
// Block: 64 i-rows, 8 waves = 4(i) x 2(v-half). Kt A-fragments hoisted
// (invariant over j). G is stored transposed [C,N] so its tile staging and
// B-fragments are pure b128 traffic. P relayout via per-wave LDS.
// ---------------------------------------------------------------------------
__global__ __launch_bounds__(256)
void attn_kernel(const _Float16* __restrict__ Q, const _Float16* __restrict__ Kt,
                 const _Float16* __restrict__ Gt, const float* __restrict__ mx,
                 const float* __restrict__ sm, _Float16* __restrict__ Oout) {
  __shared__ __attribute__((aligned(16))) _Float16 ks[64 * 128];     // 16 KB
  __shared__ __attribute__((aligned(16))) _Float16 qs[32 * 128];     // 8 KB
  __shared__ __attribute__((aligned(16))) _Float16 gs[256 * 32];     // 16 KB  G^T[v][j]
  __shared__ __attribute__((aligned(16))) _Float16 ps[8 * 16 * 32];  // 8 KB
  int i0 = blockIdx.x * 64, b = blockIdx.y;
  int tid = threadIdx.x, w = tid >> 5, lane = tid & 31;
  int iw = w & 3, vw = w >> 2;
  const _Float16* Kb = Kt + (size_t)b * NP * CI + (size_t)i0 * CI;
  const _Float16* Qb = Q + (size_t)b * NP * CI;
  const _Float16* Gb = Gt + (size_t)b * C * NP;  // [C][NP]
  {
    const uint4* src = (const uint4*)Kb;
    uint4* dst = (uint4*)ks;
    for (int i = tid; i < 64 * 128 / 8; i += 256) dst[i] = src[i];
  }
  __syncthreads();
  v16h ak[4];  // loop-invariant A fragments (this wave's 16 i-rows)
#pragma unroll
  for (int kk = 0; kk < 4; ++kk)
    ak[kk] = frag_a_rm(ks + (iw * 16) * 128 + kk * 32, 128);

  v8f zeroV = {};
  v8f accO[8];
#pragma unroll
  for (int t = 0; t < 8; ++t) accO[t] = zeroV;
  int hh = lane >> 4, nn = lane & 15;

  for (int jt = 0; jt < NP / 32; ++jt) {
    int j0 = jt * 32;
    {
      const uint4* sq = (const uint4*)(Qb + (size_t)j0 * CI);
      for (int i = tid; i < 32 * 128 / 8; i += 256) ((uint4*)qs)[i] = sq[i];
      // G^T tile: rows v, 32 halves of j each -> uint4 copies
      for (int idx = tid; idx < 256 * 32 / 8; idx += 256) {
        int v = idx >> 2, part = idx & 3;
        ((uint4*)gs)[idx] = *(const uint4*)(Gb + (size_t)v * NP + j0 + part * 8);
      }
      if (jt + 1 < NP / 32)
        __builtin_prefetch(Qb + (size_t)(j0 + 32) * CI + tid * 16, 0, 1);
    }
    __syncthreads();
    _Float16* pw = ps + w * (16 * 32);
#pragma unroll
    for (int jh = 0; jh < 2; ++jh) {
      v8f s = {};
#pragma unroll
      for (int kk = 0; kk < 4; ++kk) {
        v16h bf = frag_b_nm(qs + (jh * 16) * 128 + kk * 32, 128);
        s = wmma_f16(ak[kk], bf, s);
      }
      int j = j0 + jh * 16 + nn;
      float mxv = mx[(size_t)b * NP + j];
      float inv = 1.0f / sm[(size_t)b * NP + j];
#pragma unroll
      for (int r = 0; r < 8; ++r) {
        float p = __expf(s[r] - mxv) * inv;
        pw[(r + 8 * hh) * 32 + jh * 16 + nn] = (_Float16)p;  // D-frag -> LDS
      }
    }
#if __has_builtin(__builtin_amdgcn_s_wait_dscnt)
    __builtin_amdgcn_s_wait_dscnt(0);  // same-wave LDS relayout visibility
#endif
    v16h ap = frag_a_rm(pw, 32);  // reload as A-fragment (K = 32 j's)
#pragma unroll
    for (int t = 0; t < 8; ++t) {
      v16h bf = frag_b_nm(gs + (vw * 128 + t * 16) * 32, 32);
      accO[t] = wmma_f16(ap, bf, accO[t]);
    }
    __syncthreads();
  }
  _Float16* Ob = Oout + (size_t)b * NP * C;
#pragma unroll
  for (int t = 0; t < 8; ++t)
#pragma unroll
    for (int r = 0; r < 8; ++r) {
      int m = r + 8 * hh;
      Ob[(size_t)(i0 + iw * 16 + m) * C + vw * 128 + t * 16 + nn] =
          (_Float16)accO[t][r];
    }
}

// ---------------------------------------------------------------------------
// Kernel 4: final 1x1 conv + residual. out[b,o,n] = sum_v Wm[o,v]*O[b,n,v] + x0
// ---------------------------------------------------------------------------
__global__ __launch_bounds__(256)
void mask_kernel(const _Float16* __restrict__ Oin, const float* __restrict__ Wm,
                 const float* __restrict__ x0, float* __restrict__ out) {
  __shared__ __attribute__((aligned(16))) _Float16 os[128 * 32];
  __shared__ __attribute__((aligned(16))) _Float16 ms[128 * 32];
  int n0 = blockIdx.x * 128, o0 = blockIdx.y * 128, b = blockIdx.z;
  const _Float16* Ob = Oin + (size_t)b * NP * C;
  int tid = threadIdx.x, w = tid >> 5, lane = tid & 31;
  v8f zeroV = {};
  v8f acc[8];
#pragma unroll
  for (int t = 0; t < 8; ++t) acc[t] = zeroV;
  for (int v0 = 0; v0 < C; v0 += 32) {
    for (int idx = tid; idx < 128 * 32 / 8; idx += 256) {
      int m = idx >> 2, part = idx & 3;
      ((uint4*)os)[idx] =
          *(const uint4*)(Ob + (size_t)(n0 + m) * C + v0 + part * 8);
    }
    for (int idx = tid; idx < 128 * 8; idx += 256) {
      int n = idx >> 3, kq = idx & 7;
      float4 v = *(const float4*)(Wm + (size_t)(o0 + n) * C + v0 + kq * 4);
      union { _Float16 h[4]; uint2 u; } pk; cvt4(pk.h, v);
      *(uint2*)(ms + n * 32 + kq * 4) = pk.u;
    }
    __syncthreads();
    v16h a = frag_a_rm(os + (w * 16) * 32, 32);
#pragma unroll
    for (int t = 0; t < 8; ++t) {
      v16h bf = frag_b_nm(ms + (t * 16) * 32, 32);
      acc[t] = wmma_f16(a, bf, acc[t]);
    }
    __syncthreads();
  }
  int hh = lane >> 4, nn = lane & 15;
#pragma unroll
  for (int t = 0; t < 8; ++t)
#pragma unroll
    for (int r = 0; r < 8; ++r) {
      int m = r + 8 * hh;
      size_t idx = ((size_t)b * (2 * C) + o0 + t * 16 + nn) * NP + n0 + w * 16 + m;
      out[idx] = acc[t][r] + x0[idx];
    }
}

extern "C" void kernel_launch(void* const* d_in, const int* in_sizes, int n_in,
                              void* d_out, int out_size, void* d_ws, size_t ws_size,
                              hipStream_t stream) {
  (void)in_sizes; (void)n_in; (void)out_size; (void)ws_size;
  const float* x0     = (const float*)d_in[0];
  const float* x      = (const float*)d_in[1];
  const float* x_dsm  = (const float*)d_in[2];
  const float* w_phi  = (const float*)d_in[3];
  const float* w_theta= (const float*)d_in[4];
  const float* w_g    = (const float*)d_in[5];
  const float* w_mask = (const float*)d_in[6];
  float* out = (float*)d_out;

  char* ws = (char*)d_ws;
  size_t off = 0;
  auto alloc = [&](size_t bytes) {
    char* p = ws + off;
    off = (off + bytes + 255) & ~(size_t)255;
    return p;
  };
  _Float16* Q    = (_Float16*)alloc((size_t)Bn * NP * CI * sizeof(_Float16));
  _Float16* Kt   = (_Float16*)alloc((size_t)Bn * NP * CI * sizeof(_Float16));
  _Float16* Gt   = (_Float16*)alloc((size_t)Bn * NP * C  * sizeof(_Float16));
  _Float16* Obuf = (_Float16*)alloc((size_t)Bn * NP * C  * sizeof(_Float16));
  float*    mx   = (float*)alloc((size_t)Bn * NP * sizeof(float));
  float*    sm   = (float*)alloc((size_t)Bn * NP * sizeof(float));

  dim3 blk(256);
  // Projections: Q = phi^T [N,ci], Kt = theta [N,ci], Gt = g^T [c,N] (f16)
  proj_kernel<0><<<dim3(NP / 128, CI / 128, Bn), blk, 0, stream>>>(w_phi, x, Q, CI, C);
  proj_kernel<0><<<dim3(NP / 128, CI / 128, Bn), blk, 0, stream>>>(w_theta, x_dsm, Kt, CI, C);
  proj_kernel<1><<<dim3(NP / 128, C / 128, Bn), blk, 0, stream>>>(w_g, x0, Gt, C, 2 * C);
  // Column-softmax stats (online, recompute S)
  stats_kernel<<<dim3(NP / 128, Bn), blk, 0, stream>>>(Q, Kt, mx, sm);
  // Recompute S, apply softmax weights, accumulate P*G
  attn_kernel<<<dim3(NP / 64, Bn), blk, 0, stream>>>(Q, Kt, Gt, mx, sm, Obuf);
  // Final 1x1 conv + residual
  mask_kernel<<<dim3(NP / 128, (2 * C) / 128, Bn), blk, 0, stream>>>(Obuf, w_mask, x0, out);
}
